// ChannelAttention_7868380086682
// MI455X (gfx1250) — matrix operand order repaired and checked by reading, not compiled, not run on hardware
//
#include <hip/hip_runtime.h>
#include <hip/hip_bf16.h>

// ChannelAttention for MI455X (gfx1250), wave32 + WMMA bf16.
//   K1: scores[b,k,l] = sum_n x[b,n,k] * x[b,T(n),l]      (bf16 WMMA, f32 acc)
//   K2: row softmax over l, written transposed (S_T[j][i])
//   K3: out[b,n,j] = beta[j] * sum_i x[b,n,i]*S_T[j,i] + x[b,n,j]
// Workspace: 32 MB of d_ws (16 MB scores + 16 MB S_T).
//
// LDS tiles use an 80-byte row stride so every WMMA fragment is exactly two
// 16B-aligned chunks per lane -> ds_load_b128. K1's transposed fill adds a
// chunk-level XOR swizzle ((row>>4)&3) to avoid row aliasing conflicts.

typedef __attribute__((ext_vector_type(16))) __bf16 v16bf;
typedef __attribute__((ext_vector_type(8)))  __bf16 v8bf;
typedef __attribute__((ext_vector_type(2)))  __bf16 bf16x2;
typedef __attribute__((ext_vector_type(8)))  float  v8f;

#define C_DIM 512
#define NSP   4096      // 64*64 spatial
#define BATCH 16
#define KSTEP 32        // K-chunk per WMMA step (16x16x32 bf16)
#define LSTRIDE 40      // LDS row stride in bf16 units: 80B rows, 16B-aligned chunks

// Two f32 -> packed bf16x2 (clang lowers the fptrunc pair to the packed cvt).
__device__ __forceinline__ unsigned pack_bf2(float lo, float hi) {
  bf16x2 p;
  p.x = (__bf16)lo;
  p.y = (__bf16)hi;
  return __builtin_bit_cast(unsigned, p);
}

// A fragment: 16x32 (MxK) bf16, ISA wave32 layout.
// lane m = lane&15; lanes 0-15 hold K {0..7,16..23}, lanes 16-31 hold K {8..15,24..31}.
// Row = 4 chunks of 8 bf16; SWZ applies chunk ^= (row>>4)&3.
template <bool SWZ>
__device__ __forceinline__ v16bf load_a_frag(const __bf16* base, int row0, int lane) {
  const int m  = row0 + (lane & 15);
  const int sw = SWZ ? ((m >> 4) & 3) : 0;
  const int c0 = (lane >> 4) & 1;                 // chunk 0 or 1 (K 0..7 / 8..15)
  const __bf16* rp = base + (size_t)m * LSTRIDE;
  v8bf lo = *(const v8bf*)(rp + ((c0 ^ sw) * 8));        // K {0..7 | 8..15}
  v8bf hi = *(const v8bf*)(rp + (((c0 + 2) ^ sw) * 8));  // K {16..23 | 24..31}
  return __builtin_shufflevector(lo, hi, 0, 1, 2, 3, 4, 5, 6, 7,
                                 8, 9, 10, 11, 12, 13, 14, 15);
}

// B fragment: 32x16 (KxN) bf16. LDS row = N column, K contiguous.
// lane n = lane&15; lanes 0-15 hold K {0..7,16..23}, lanes 16-31 hold K {8..15,24..31}.
template <bool SWZ>
__device__ __forceinline__ v16bf load_b_frag(const __bf16* base, int col0, int lane) {
  const int n  = col0 + (lane & 15);
  const int sw = SWZ ? ((n >> 4) & 3) : 0;
  const int c0 = (lane >> 4) & 1;                 // chunk 0 or 1 (K 0..7 / 8..15)
  const __bf16* rp = base + (size_t)n * LSTRIDE;
  v8bf lo = *(const v8bf*)(rp + ((c0 ^ sw) * 8));        // K {0..7 | 8..15}
  v8bf hi = *(const v8bf*)(rp + (((c0 + 2) ^ sw) * 8));  // K {16..23 | 24..31}
  return __builtin_shufflevector(lo, hi, 0, 1, 2, 3, 4, 5, 6, 7,
                                 8, 9, 10, 11, 12, 13, 14, 15);
}

// Swizzled address of a K-pair word in a transposed-fill row (K1).
__device__ __forceinline__ unsigned* pair_ptr(__bf16* base, int row, int np) {
  const int chunk = (np >> 2) ^ ((row >> 4) & 3);
  return (unsigned*)(base + (size_t)row * LSTRIDE + chunk * 8 + (np & 3) * 2);
}

// ---------------- Kernel 1: gram matmul  scores = A^T * A' -------------------
__global__ __launch_bounds__(256)
void gram_wmma_kernel(const float* __restrict__ x, float* __restrict__ scores) {
  __shared__ __align__(16) __bf16 ldsA[2][128][LSTRIDE];   // [M=channel][K=spatial]
  __shared__ __align__(16) __bf16 ldsB[2][128][LSTRIDE];   // [N=channel][K=spatial]

  const int tid  = threadIdx.x;
  const int lane = tid & 31;
  const int wave = tid >> 5;
  const int wM   = wave & 3;               // 4 waves along M
  const int wN   = wave >> 2;              // 2 waves along N
  const int b    = blockIdx.z;
  const int m0   = blockIdx.x * 128;       // scores k (channel) tile
  const int n0   = blockIdx.y * 128;       // scores l (channel) tile
  const float* xb = x + (size_t)b * NSP * C_DIM;

  v8f acc[2][4] = {};

  // Cooperative fill of one K-chunk: each thread handles a spatial PAIR so the
  // transposed write is a packed 32-bit word (low=even K, high=odd K).
  auto load_tiles = [&](int buf, int k0) {
#pragma unroll
    for (int it = 0; it < 2; ++it) {
      int q   = tid + it * 256;            // 0..511
      int np  = q >> 5;                    // spatial pair index (0..15)
      int c4  = q & 31;                    // float4 slot over 128 channels
      int ns0 = k0 + 2 * np;
      int ns1 = ns0 + 1;
      // A: x[b, spatial, m0+channel]  (coalesced over channels)
      const float4 a0 = *(const float4*)(xb + (size_t)ns0 * C_DIM + m0 + c4 * 4);
      const float4 a1 = *(const float4*)(xb + (size_t)ns1 * C_DIM + m0 + c4 * 4);
      if (k0 + KSTEP < NSP)
        __builtin_prefetch(xb + (size_t)(ns0 + KSTEP) * C_DIM + m0 + c4 * 4, 0, 1);
      // B: x[b, T(spatial), n0+channel]
      int t0 = ((ns0 & 63) << 6) | (ns0 >> 6);
      int t1 = ((ns1 & 63) << 6) | (ns1 >> 6);
      const float4 b0 = *(const float4*)(xb + (size_t)t0 * C_DIM + n0 + c4 * 4);
      const float4 b1 = *(const float4*)(xb + (size_t)t1 * C_DIM + n0 + c4 * 4);
      const int c = c4 * 4;
      *pair_ptr(&ldsA[buf][0][0], c + 0, np) = pack_bf2(a0.x, a1.x);
      *pair_ptr(&ldsA[buf][0][0], c + 1, np) = pack_bf2(a0.y, a1.y);
      *pair_ptr(&ldsA[buf][0][0], c + 2, np) = pack_bf2(a0.z, a1.z);
      *pair_ptr(&ldsA[buf][0][0], c + 3, np) = pack_bf2(a0.w, a1.w);
      *pair_ptr(&ldsB[buf][0][0], c + 0, np) = pack_bf2(b0.x, b1.x);
      *pair_ptr(&ldsB[buf][0][0], c + 1, np) = pack_bf2(b0.y, b1.y);
      *pair_ptr(&ldsB[buf][0][0], c + 2, np) = pack_bf2(b0.z, b1.z);
      *pair_ptr(&ldsB[buf][0][0], c + 3, np) = pack_bf2(b0.w, b1.w);
    }
  };

  load_tiles(0, 0);
  __syncthreads();

  const int KT = NSP / KSTEP;              // 128 K-steps
  for (int kt = 0; kt < KT; ++kt) {
    const int cur = kt & 1;
    if (kt + 1 < KT) load_tiles(cur ^ 1, (kt + 1) * KSTEP);

    v16bf afrag[2], bfrag[4];
#pragma unroll
    for (int r = 0; r < 2; ++r)
      afrag[r] = load_a_frag<true>(&ldsA[cur][0][0], wM * 32 + r * 16, lane);
#pragma unroll
    for (int s = 0; s < 4; ++s)
      bfrag[s] = load_b_frag<true>(&ldsB[cur][0][0], wN * 64 + s * 16, lane);

#pragma unroll
    for (int r = 0; r < 2; ++r)
#pragma unroll
      for (int s = 0; s < 4; ++s)
        acc[r][s] = __builtin_amdgcn_wmma_f32_16x16x32_bf16(
            false, afrag[r], false, bfrag[s], (short)0, acc[r][s], false, false);

    __syncthreads();
  }

  // Store f32 scores. C/D layout: lane col = lane&15, VGPR v row = v + (lane>>4)*8.
  float* Sb = scores + (size_t)b * C_DIM * C_DIM;
  const int colL = lane & 15;
  const int hi8  = ((lane >> 4) & 1) * 8;
#pragma unroll
  for (int r = 0; r < 2; ++r) {
    const int mrow = m0 + wM * 32 + r * 16 + hi8;
#pragma unroll
    for (int s = 0; s < 4; ++s) {
      const int ncol = n0 + wN * 64 + s * 16 + colL;
#pragma unroll
      for (int v = 0; v < 8; ++v)
        Sb[(size_t)(mrow + v) * C_DIM + ncol] = acc[r][s][v];
    }
  }
}

// ---------------- Kernel 2: row softmax + transpose --------------------------
__global__ __launch_bounds__(256)
void softmax_T_kernel(const float* __restrict__ scores, float* __restrict__ st) {
  const int lane = threadIdx.x & 31;
  const int wave = threadIdx.x >> 5;
  const int row  = blockIdx.x * 8 + wave;      // global row (b*512 + i)
  const int b    = row >> 9;
  const int i    = row & 511;
  const float* rp = scores + (size_t)row * C_DIM;

  float vals[16];
  float mx = -3.402823466e+38f;
#pragma unroll
  for (int e = 0; e < 16; ++e) {
    vals[e] = rp[lane + e * 32];
    mx = fmaxf(mx, vals[e]);
  }
#pragma unroll
  for (int o = 16; o > 0; o >>= 1) mx = fmaxf(mx, __shfl_xor(mx, o, 32));

  float sum = 0.f;
#pragma unroll
  for (int e = 0; e < 16; ++e) {
    vals[e] = __expf(vals[e] - mx);
    sum += vals[e];
  }
#pragma unroll
  for (int o = 16; o > 0; o >>= 1) sum += __shfl_xor(sum, o, 32);
  const float inv = 1.0f / sum;

  // write transposed: S_T[b][j][i]
  float* op = st + (size_t)b * C_DIM * C_DIM + i;
#pragma unroll
  for (int e = 0; e < 16; ++e)
    op[(size_t)(lane + e * 32) * C_DIM] = vals[e] * inv;
}

// ---------------- Kernel 3: out = beta * (X * S) + x -------------------------
__global__ __launch_bounds__(256)
void apply_wmma_kernel(const float* __restrict__ x, const float* __restrict__ st,
                       const float* __restrict__ beta, float* __restrict__ out) {
  __shared__ __align__(16) __bf16 ldsA[2][128][LSTRIDE];   // X: [M=spatial][K=i]
  __shared__ __align__(16) __bf16 ldsB[2][128][LSTRIDE];   // S_T: [N=j][K=i]

  const int tid  = threadIdx.x;
  const int lane = tid & 31;
  const int wave = tid >> 5;
  const int wM   = wave & 3;
  const int wN   = wave >> 2;
  const int b    = blockIdx.z;
  const int m0   = blockIdx.x * 128;       // spatial tile (0..4095)
  const int n0   = blockIdx.y * 128;       // output channel tile
  const float* xb = x  + (size_t)b * NSP * C_DIM;
  const float* sb = st + (size_t)b * C_DIM * C_DIM;

  v8f acc[2][4] = {};

  // Natural-layout fill: each thread converts one 16B LDS chunk (8 floats) and
  // writes it with a single ds_store_b128.
  auto load_tiles = [&](int buf, int k0) {
#pragma unroll
    for (int it = 0; it < 2; ++it) {
      int q  = tid + it * 256;             // 0..511
      int m  = q >> 2;                     // row 0..127
      int ch = q & 3;                      // 16B chunk within 32-wide K slab
      const float* pa = xb + (size_t)(m0 + m) * C_DIM + k0 + ch * 8;
      const float* pb = sb + (size_t)(n0 + m) * C_DIM + k0 + ch * 8;
      const float4 a0 = *(const float4*)(pa);
      const float4 a1 = *(const float4*)(pa + 4);
      const float4 b0 = *(const float4*)(pb);
      const float4 b1 = *(const float4*)(pb + 4);
      uint4 wa, wb;
      wa.x = pack_bf2(a0.x, a0.y); wa.y = pack_bf2(a0.z, a0.w);
      wa.z = pack_bf2(a1.x, a1.y); wa.w = pack_bf2(a1.z, a1.w);
      wb.x = pack_bf2(b0.x, b0.y); wb.y = pack_bf2(b0.z, b0.w);
      wb.z = pack_bf2(b1.x, b1.y); wb.w = pack_bf2(b1.z, b1.w);
      *(uint4*)&ldsA[buf][m][ch * 8] = wa;   // 16B-aligned (stride 80B)
      *(uint4*)&ldsB[buf][m][ch * 8] = wb;
    }
  };

  load_tiles(0, 0);
  __syncthreads();

  const int KT = C_DIM / KSTEP;            // 16 K-steps
  for (int kt = 0; kt < KT; ++kt) {
    const int cur = kt & 1;
    if (kt + 1 < KT) load_tiles(cur ^ 1, (kt + 1) * KSTEP);

    v16bf afrag[2], bfrag[4];
#pragma unroll
    for (int r = 0; r < 2; ++r)
      afrag[r] = load_a_frag<false>(&ldsA[cur][0][0], wM * 32 + r * 16, lane);
#pragma unroll
    for (int s = 0; s < 4; ++s)
      bfrag[s] = load_b_frag<false>(&ldsB[cur][0][0], wN * 64 + s * 16, lane);

#pragma unroll
    for (int r = 0; r < 2; ++r)
#pragma unroll
      for (int s = 0; s < 4; ++s)
        acc[r][s] = __builtin_amdgcn_wmma_f32_16x16x32_bf16(
            false, afrag[r], false, bfrag[s], (short)0, acc[r][s], false, false);

    __syncthreads();
  }

  // Fused epilogue: out = beta[j]*acc + x
  float* ob = out + (size_t)b * NSP * C_DIM;
  const int colL = lane & 15;
  const int hi8  = ((lane >> 4) & 1) * 8;
#pragma unroll
  for (int r = 0; r < 2; ++r) {
    const int mrow = m0 + wM * 32 + r * 16 + hi8;
#pragma unroll
    for (int s = 0; s < 4; ++s) {
      const int jc = n0 + wN * 64 + s * 16 + colL;
      const float bt = beta[jc];
#pragma unroll
      for (int v = 0; v < 8; ++v) {
        const size_t idx = (size_t)(mrow + v) * C_DIM + jc;
        ob[idx] = bt * acc[r][s][v] + xb[idx];
      }
    }
  }
}

extern "C" void kernel_launch(void* const* d_in, const int* in_sizes, int n_in,
                              void* d_out, int out_size, void* d_ws, size_t ws_size,
                              hipStream_t stream) {
  (void)in_sizes; (void)n_in; (void)out_size; (void)ws_size;
  const float* x    = (const float*)d_in[0];
  const float* beta = (const float*)d_in[1];
  float* out = (float*)d_out;

  float* scores = (float*)d_ws;                                  // 16 MB
  float* st     = scores + (size_t)BATCH * C_DIM * C_DIM;        // next 16 MB

  // K1: 512x512 gram per batch; 128x128 tiles -> grid (4,4,16)
  gram_wmma_kernel<<<dim3(4, 4, BATCH), 256, 0, stream>>>(x, scores);
  // K2: 8192 rows, 8 rows (waves) per block
  softmax_T_kernel<<<dim3(BATCH * C_DIM / 8), 256, 0, stream>>>(scores, st);
  // K3: 4096x512 per batch; 128x128 tiles -> grid (32,4,16)
  apply_wmma_kernel<<<dim3(32, 4, BATCH), 256, 0, stream>>>(x, st, beta, out);
}
